// ZPEEnhancedFiveDimensionalHypercubeNN_52218212385241
// MI455X (gfx1250) — matrix operand order repaired and checked by reading, not compiled, not run on hardware
//
#include <hip/hip_runtime.h>

typedef unsigned int  u32;
typedef unsigned short u16;
typedef unsigned long long u64;
typedef __attribute__((ext_vector_type(4)))  u32    u32x4;
typedef __attribute__((ext_vector_type(8)))  float  v8f;
typedef __attribute__((ext_vector_type(16))) __bf16 v16bf;

#define BB 2048
#define DD 512
#define HH 512
#define OO 512
#define LL 10
#define NVV 32
#define NFF 5

__device__ __forceinline__ u16 f2bf(float f) {
  u32 u = __float_as_uint(f);
  u32 r = (u + 0x7FFFu + ((u >> 16) & 1u)) >> 16;   // round-to-nearest-even
  return (u16)r;
}

union Frag { u32x4 q[2]; v16bf v; };

// ---------------------------------------------------------------------------
// bf16 WMMA GEMM with async global->LDS double buffering:
//   Out[M,N] = act(A[M,K](bf16) @ W[N,K](bf16)^T + bias[N])
//   ACT: 0 = none -> f32 out, 1 = relu -> f32 out, 2 = tanh -> bf16 out
// Macro tile 128x64x32, 256 threads (8 wave32), wave grid 4x2, 2x2 frags/wave.
// Tiles are moved with GLOBAL_LOAD_ASYNC_TO_LDS_B128 (ASYNCcnt) and the
// compute pipeline double-buffers LDS (2x8KB A, 2x4KB B).
// ---------------------------------------------------------------------------
#define TBM 128
#define TBN 64
#define TBK 32

template<int ACT>
__global__ void __launch_bounds__(256)
gemm_wmma(const u16* __restrict__ A, const u16* __restrict__ W,
          const float* __restrict__ bias, void* __restrict__ Out,
          int K, int ldo,
          long strideA, long strideW, long strideBias, long strideOut)
{
  __shared__ u32 shA[2][TBM * 16];   // 2 x (128 rows x 32 bf16) = 2 x 8 KB
  __shared__ u32 shB[2][TBN * 16];   // 2 x ( 64 rows x 32 bf16) = 2 x 4 KB

  const int tid  = threadIdx.x;
  const int lane = tid & 31;
  const int wid  = tid >> 5;
  const int wm   = wid >> 1;           // 0..3
  const int wn   = wid & 1;            // 0..1
  const int bm0  = blockIdx.y * TBM;
  const int bn0  = blockIdx.x * TBN;
  const int z    = blockIdx.z;

  const char*  Abase = (const char*)(A + (size_t)z * strideA);
  const char*  Wbase = (const char*)(W + (size_t)z * strideW + (size_t)bn0 * K);
  const float* bp    = bias + (size_t)z * strideBias + bn0;

  // per-thread async-copy offsets (16B chunks); row bytes = K*2
  const u32 KB    = (u32)K * 2u;
  const u32 aOff0 = (u32)(bm0 + (tid >> 2))        * KB + (u32)(tid & 3) * 16u;
  const u32 aOff1 = (u32)(bm0 + ((tid + 256) >> 2)) * KB + (u32)(tid & 3) * 16u;
  const u32 bOff  = (u32)(tid >> 2)                * KB + (u32)(tid & 3) * 16u;
  const u32 ldsA0 = (u32)(size_t)(&shA[0][0]);
  const u32 ldsB0 = (u32)(size_t)(&shB[0][0]);
  const u32 ldsAme = ldsA0 + (u32)tid * 16u;            // chunk tid
  const u32 ldsBme = ldsB0 + (u32)tid * 16u;

  auto issue_tile = [&](int kt, int bs) {
    u32 ka = (u32)kt * 64u;                             // 32 bf16 = 64 bytes
    u32 la0 = ldsAme + (u32)bs * 8192u;
    u32 la1 = la0 + 4096u;                              // chunk tid+256
    u32 lb  = ldsBme + (u32)bs * 4096u;
    u32 ga0 = aOff0 + ka, ga1 = aOff1 + ka, gb = bOff + ka;
    asm volatile("global_load_async_to_lds_b128 %0, %1, %2"
                 :: "v"(la0), "v"(ga0), "s"(Abase) : "memory");
    asm volatile("global_load_async_to_lds_b128 %0, %1, %2"
                 :: "v"(la1), "v"(ga1), "s"(Abase) : "memory");
    asm volatile("global_load_async_to_lds_b128 %0, %1, %2"
                 :: "v"(lb), "v"(gb), "s"(Wbase) : "memory");
  };

  v8f acc[2][2] = {};
  const int nK = K / TBK;

  issue_tile(0, 0);
  for (int kt = 0; kt < nK; ++kt) {
    const int bs = kt & 1;
    asm volatile("s_wait_asynccnt 0x0" ::: "memory");   // own tile-kt copies done
    __syncthreads();                                    // everyone's copies visible
    if (kt + 1 < nK) issue_tile(kt + 1, bs ^ 1);        // overlap next copy w/ math

    // ---- fragment loads per documented wave32 layouts ----
    const int m  = lane & 15;
    const int kh = lane >> 4;
    Frag fa[2], fb[2];
#pragma unroll
    for (int f = 0; f < 2; ++f) {
      int ra = (wm * 32 + f * 16 + m) * 16;
      fa[f].q[0] = *(const u32x4*)&shA[bs][ra + kh * 4];       // K 0..7 / 8..15
      fa[f].q[1] = *(const u32x4*)&shA[bs][ra + 8 + kh * 4];   // K 16..23 / 24..31
      int rb = (wn * 32 + f * 16 + m) * 16;
      fb[f].q[0] = *(const u32x4*)&shB[bs][rb + kh * 4];       // K 0..7 / 8..15
      fb[f].q[1] = *(const u32x4*)&shB[bs][rb + 8 + kh * 4];   // K 16..23 / 24..31
    }
#pragma unroll
    for (int i = 0; i < 2; ++i)
#pragma unroll
      for (int j = 0; j < 2; ++j)
        acc[i][j] = __builtin_amdgcn_wmma_f32_16x16x32_bf16(
            false, fa[i].v, false, fb[j].v, (short)0, acc[i][j], false, false);
    __syncthreads();   // all waves done reading buffer bs before it is refilled
  }

  // ---- epilogue: C layout lane n = lane&15, M = (lane>=16?8:0)+r ----
  const int n  = lane & 15;
  const int mb = (lane >> 4) * 8;
#pragma unroll
  for (int i = 0; i < 2; ++i)
#pragma unroll
    for (int j = 0; j < 2; ++j)
#pragma unroll
      for (int r = 0; r < 8; ++r) {
        int gm = bm0 + wm * 32 + i * 16 + mb + r;
        int ln = wn * 32 + j * 16 + n;          // local N within block tile
        float v = acc[i][j][r] + bp[ln];
        size_t off = (size_t)gm * ldo + (bn0 + ln) + (size_t)z * strideOut;
        if constexpr (ACT == 1) v = fmaxf(v, 0.0f);
        if constexpr (ACT == 2) {
          ((u16*)Out)[off] = f2bf(tanhf(v));
        } else {
          ((float*)Out)[off] = v;
        }
      }
}

// ---------------------------------------------------------------------------
// Small helper kernels
// ---------------------------------------------------------------------------
__global__ void f2bf_kernel(const float* __restrict__ s, u16* __restrict__ d, long n) {
  for (long i = (long)blockIdx.x * blockDim.x + threadIdx.x; i < n;
       i += (long)gridDim.x * blockDim.x)
    d[i] = f2bf(s[i]);
}

// wvb[v,o,i] = bf16(Wv[v,o,i] * combined[v,i])  (folds the per-vertex column
// scaling of the A operand into the weights: (zin*c)@Wv^T == zin@(Wv*c)^T)
__global__ void wvscale_kernel(const float* __restrict__ Wv,
                               const float* __restrict__ combined,
                               u16* __restrict__ out, long n) {
  for (long i = (long)blockIdx.x * blockDim.x + threadIdx.x; i < n;
       i += (long)gridDim.x * blockDim.x) {
    int col = (int)(i & (HH - 1));
    int v   = (int)(i >> 18);                 // H*H = 2^18
    out[i] = f2bf(Wv[i] * combined[v * HH + col]);
  }
}

__global__ void zero2_kernel(float* a) { if (threadIdx.x < 2) a[threadIdx.x] = 0.0f; }

__global__ void abssum_kernel(const float* __restrict__ x, long n, float* acc) {
  __shared__ float sh[256];
  float s = 0.0f;
  for (long i = (long)blockIdx.x * blockDim.x + threadIdx.x; i < n;
       i += (long)gridDim.x * blockDim.x)
    s += fabsf(x[i]);
  sh[threadIdx.x] = s;
  __syncthreads();
  for (int o = 128; o > 0; o >>= 1) {
    if (threadIdx.x < o) sh[threadIdx.x] += sh[threadIdx.x + o];
    __syncthreads();
  }
  if (threadIdx.x == 0) atomicAdd(acc, sh[0]);
}

// new_zpe[l] = clip(0.9*zpe[l] + 0.1*(1 + tanh(mean|.|*0.3)*0.2), 0.5, 1.5)
__global__ void zpe_kernel(const float* __restrict__ zpe_in, const float* acc,
                           float inv_n, float* __restrict__ zpe_out) {
  int l = threadIdx.x;
  if (l < LL) {
    float pert = tanhf(acc[0] * inv_n * 0.3f);
    float v = 0.9f * zpe_in[l] + 0.1f * (1.0f + pert * 0.2f);
    zpe_out[l] = fminf(fmaxf(v, 0.5f), 1.5f);
  }
}

// zin = ti * izpe_tiled   (f32 copy for column means, bf16 copy for GEMMs)
__global__ void zin_kernel(const float* __restrict__ ti, const float* __restrict__ izpe,
                           float* __restrict__ zin, u16* __restrict__ zinb, long n) {
  for (long i = (long)blockIdx.x * blockDim.x + threadIdx.x; i < n;
       i += (long)gridDim.x * blockDim.x) {
    float v = ti[i] * izpe[((int)(i & (HH - 1))) % LL];
    zin[i]  = v;
    zinb[i] = f2bf(v);
  }
}

__global__ void colmean_kernel(const float* __restrict__ zin, float* __restrict__ bmean) {
  __shared__ float sh[256];
  int h = blockIdx.x;
  float s = 0.0f;
  for (int b = threadIdx.x; b < BB; b += 256) s += zin[(size_t)b * HH + h];
  sh[threadIdx.x] = s;
  __syncthreads();
  for (int o = 128; o > 0; o >>= 1) {
    if (threadIdx.x < o) sh[threadIdx.x] += sh[threadIdx.x + o];
    __syncthreads();
  }
  if (threadIdx.x == 0) bmean[h] = sh[0] / (float)BB;
}

// zmod -> new_flows -> combined[v,h] = prod_f(coords[v,f]==1 ? nf[v,f,h%L] : 1)
__global__ void vertex_kernel(const float* __restrict__ Wm, const float* __restrict__ bm,
                              const float* __restrict__ flows, const int* __restrict__ coords,
                              const float* __restrict__ bmean, float* __restrict__ combined) {
  __shared__ float nf[NVV * NFF * LL];   // 1600 floats
  __shared__ float bsh[HH];
  int tid = threadIdx.x;
  for (int i = tid; i < HH; i += 256) bsh[i] = bmean[i];
  __syncthreads();
  for (int p = tid; p < NVV * NFF; p += 256) {
    const float* w = Wm + (size_t)p * HH;
    float s = 0.0f;
    for (int h = 0; h < HH; ++h) s += w[h] * bsh[h];
    float zm   = 1.0f / (1.0f + expf(-(s + bm[p])));
    float pert = tanhf(zm * 0.3f);
    float t    = 1.0f + pert * 0.2f;
    for (int l = 0; l < LL; ++l) {
      float v = 0.9f * flows[p * LL + l] + 0.1f * t;
      nf[p * LL + l] = fminf(fmaxf(v, 0.1f), 2.0f);
    }
  }
  __syncthreads();
  for (int i = tid; i < NVV * HH; i += 256) {
    int v = i >> 9, h = i & (HH - 1), l = h % LL;
    float prod = 1.0f;
    for (int f = 0; f < NFF; ++f)
      if (coords[v * NFF + f] == 1) prod *= nf[(v * NFF + f) * LL + l];
    combined[i] = prod;
  }
}

// LayerNorm (biased var) + relu + bf16 convert; one block per row
__global__ void ln_relu_kernel(const float* __restrict__ Y, const float* __restrict__ g,
                               const float* __restrict__ be, u16* __restrict__ out, int RL) {
  __shared__ float s1[256], s2[256];
  int row = blockIdx.x;
  const float* y = Y + (size_t)row * RL;
  float a = 0.0f, b = 0.0f;
  for (int c = threadIdx.x; c < RL; c += 256) {
    float v = y[c];
    a += v; b += v * v;
  }
  s1[threadIdx.x] = a; s2[threadIdx.x] = b;
  __syncthreads();
  for (int o = 128; o > 0; o >>= 1) {
    if (threadIdx.x < o) { s1[threadIdx.x] += s1[threadIdx.x + o];
                           s2[threadIdx.x] += s2[threadIdx.x + o]; }
    __syncthreads();
  }
  float mean = s1[0] / (float)RL;
  float var  = s2[0] / (float)RL - mean * mean;
  float rstd = rsqrtf(var + 1e-5f);
  for (int c = threadIdx.x; c < RL; c += 256) {
    float v = (y[c] - mean) * rstd * g[c] + be[c];
    out[(size_t)row * RL + c] = f2bf(fmaxf(v, 0.0f));
  }
}

__global__ void scale_out_kernel(const float* __restrict__ cs, const float* __restrict__ ozpe,
                                 float* __restrict__ out, long n) {
  for (long i = (long)blockIdx.x * blockDim.x + threadIdx.x; i < n;
       i += (long)gridDim.x * blockDim.x)
    out[i] = cs[i] * ozpe[((int)(i & (OO - 1))) % LL];
}

// ---------------------------------------------------------------------------
extern "C" void kernel_launch(void* const* d_in, const int* in_sizes, int n_in,
                              void* d_out, int out_size, void* d_ws, size_t ws_size,
                              hipStream_t stream) {
  (void)in_sizes; (void)n_in; (void)out_size; (void)ws_size;
  const float* x     = (const float*)d_in[0];
  const float* Wi    = (const float*)d_in[1];
  const float* bi    = (const float*)d_in[2];
  const float* izpe0 = (const float*)d_in[3];
  // d_in[4..7] router, d_in[10,11,14] gates: dead code w.r.t. the output
  const float* Wv    = (const float*)d_in[8];
  const float* bv    = (const float*)d_in[9];
  const float* Wm    = (const float*)d_in[12];
  const float* bm    = (const float*)d_in[13];
  const float* flows = (const float*)d_in[15];
  const float* Wa1   = (const float*)d_in[16];
  const float* ba1   = (const float*)d_in[17];
  const float* g1    = (const float*)d_in[18];
  const float* be1   = (const float*)d_in[19];
  const float* Wa2   = (const float*)d_in[20];
  const float* ba2   = (const float*)d_in[21];
  const float* g2    = (const float*)d_in[22];
  const float* be2   = (const float*)d_in[23];
  const float* Wf    = (const float*)d_in[24];
  const float* bf_   = (const float*)d_in[25];
  const float* ozpe0 = (const float*)d_in[26];
  const int*   coords= (const int*)d_in[27];
  float* out = (float*)d_out;

  char* cur = (char*)d_ws;
  auto take = [&](size_t bytes) -> char* {
    char* r = cur; cur += (bytes + 255) & ~(size_t)255; return r;
  };
  u16*   xb       = (u16*)  take((size_t)BB * DD * 2);
  u16*   wib      = (u16*)  take((size_t)HH * DD * 2);
  u16*   wvb      = (u16*)  take((size_t)NVV * HH * HH * 2);
  u16*   wa1b     = (u16*)  take((size_t)4 * HH * NVV * HH * 2);
  u16*   wa2b     = (u16*)  take((size_t)HH * 4 * HH * 2);
  u16*   wfb      = (u16*)  take((size_t)OO * HH * 2);
  float* ti       = (float*)take((size_t)BB * HH * 4);
  float* zin      = (float*)take((size_t)BB * HH * 4);
  u16*   zinb     = (u16*)  take((size_t)BB * HH * 2);
  float* bmean    = (float*)take((size_t)HH * 4);
  float* combined = (float*)take((size_t)NVV * HH * 4);
  u16*   flatb    = (u16*)  take((size_t)BB * NVV * HH * 2);
  float* y1       = (float*)take((size_t)BB * 4 * HH * 4);
  u16*   h1b      = (u16*)  take((size_t)BB * 4 * HH * 2);
  float* y2       = (float*)take((size_t)BB * HH * 4);
  u16*   h2b      = (u16*)  take((size_t)BB * HH * 2);
  float* cs       = (float*)take((size_t)BB * OO * 4);
  float* acc      = (float*)take(64);   // acc[0]=input chain, acc[1]=output chain
  float* izpe     = (float*)take(64);
  float* ozpe     = (float*)take(64);

  auto conv = [&](const float* s, u16* d, long n) {
    int blocks = (int)((n + 2047) / 2048);
    f2bf_kernel<<<blocks, 256, 0, stream>>>(s, d, n);
  };
  // one-time-per-call conversions (cheap vs the GEMMs)
  conv(x,   xb,   (long)BB * DD);
  conv(Wi,  wib,  (long)HH * DD);
  conv(Wa1, wa1b, (long)4 * HH * NVV * HH);
  conv(Wa2, wa2b, (long)HH * 4 * HH);
  conv(Wf,  wfb,  (long)OO * HH);
  zero2_kernel<<<1, 32, 0, stream>>>(acc);

  // 1) ti = relu(x @ Wi^T + bi)   [B,H]
  gemm_wmma<1><<<dim3(HH / TBN, BB / TBM, 1), 256, 0, stream>>>(
      xb, wib, bi, ti, DD, HH, 0, 0, 0, 0);

  // 2) input ZPE chain -> zin (f32 + bf16)
  abssum_kernel<<<1024, 256, 0, stream>>>(ti, (long)BB * HH, acc + 0);
  zpe_kernel<<<1, 32, 0, stream>>>(izpe0, acc + 0, 1.0f / ((float)BB * HH), izpe);
  zin_kernel<<<2048, 256, 0, stream>>>(ti, izpe, zin, zinb, (long)BB * HH);

  // 3) flows -> combined ; fold combined into bf16 Wv
  colmean_kernel<<<HH, 256, 0, stream>>>(zin, bmean);
  vertex_kernel<<<1, 256, 0, stream>>>(Wm, bm, flows, coords, bmean, combined);
  wvscale_kernel<<<4096, 256, 0, stream>>>(Wv, combined, wvb, (long)NVV * HH * HH);

  // 4) batched: flat[b, v*H+o] = tanh(zin @ (Wv[v]*c_v)^T + bv[v])
  gemm_wmma<2><<<dim3(HH / TBN, BB / TBM, NVV), 256, 0, stream>>>(
      zinb, wvb, bv, flatb, HH, NVV * HH,
      /*strideA=*/0, /*strideW=*/(long)HH * HH, /*strideBias=*/HH,
      /*strideOut=*/HH);

  // 5) y1 = flat @ Wa1^T + ba1 ; h1 = relu(LN(y1))
  gemm_wmma<0><<<dim3(4 * HH / TBN, BB / TBM, 1), 256, 0, stream>>>(
      flatb, wa1b, ba1, y1, NVV * HH, 4 * HH, 0, 0, 0, 0);
  ln_relu_kernel<<<BB, 256, 0, stream>>>(y1, g1, be1, h1b, 4 * HH);

  // 6) y2 = h1 @ Wa2^T + ba2 ; h2 = relu(LN(y2))
  gemm_wmma<0><<<dim3(HH / TBN, BB / TBM, 1), 256, 0, stream>>>(
      h1b, wa2b, ba2, y2, 4 * HH, HH, 0, 0, 0, 0);
  ln_relu_kernel<<<BB, 256, 0, stream>>>(y2, g2, be2, h2b, HH);

  // 7) cs = h2 @ Wf^T + bf
  gemm_wmma<0><<<dim3(OO / TBN, BB / TBM, 1), 256, 0, stream>>>(
      h2b, wfb, bf_, cs, HH, OO, 0, 0, 0, 0);

  // 8) output ZPE chain -> out
  abssum_kernel<<<1024, 256, 0, stream>>>(cs, (long)BB * OO, acc + 1);
  zpe_kernel<<<1, 32, 0, stream>>>(ozpe0, acc + 1, 1.0f / ((float)BB * OO), ozpe);
  scale_out_kernel<<<2048, 256, 0, stream>>>(cs, ozpe, out, (long)BB * OO);
}
